// MultiHead_7842610282769
// MI455X (gfx1250) — matrix operand order repaired and checked by reading, not compiled, not run on hardware
//
#include <hip/hip_runtime.h>
#include <hip/hip_bf16.h>

typedef __bf16 bf16_t;
typedef bf16_t v16bf __attribute__((ext_vector_type(16)));
typedef bf16_t v8bf  __attribute__((ext_vector_type(8)));
typedef float  v8f   __attribute__((ext_vector_type(8)));

#define BB 2
#define TT 2048
#define CC 1024
#define HH 16
#define DD 64

__device__ __forceinline__ bf16_t f2bf(float f) {
  union { float f; unsigned u; } a; a.f = f;
  unsigned u = a.u;
  unsigned r = u + 0x7FFFu + ((u >> 16) & 1u);   // round-to-nearest-even
  union { unsigned short s; bf16_t b; } o;
  o.s = (unsigned short)(r >> 16);
  return o.b;
}

__device__ __forceinline__ v8f wmma_bf16(v16bf a, v16bf b, v8f c) {
  // D = A(16x32 bf16) * B(32x16 bf16) + C(16x16 f32)
  return __builtin_amdgcn_wmma_f32_16x16x32_bf16(false, a, false, b, (short)0, c,
                                                 false, false);
}

// A-fragment (16x32, row-major A): lane<16 -> row m0+lane, K 0..15;
// lane>=16 -> row m0+lane-16, K 16..31
__device__ __forceinline__ v16bf load_a_frag(const bf16_t* __restrict__ A, int lda,
                                             int row0, int k0, int lane) {
  int r    = row0 + (lane & 15);
  int koff = (lane < 16) ? 0 : 16;
  const bf16_t* p = A + (size_t)r * lda + k0 + koff;
  union { v16bf f; v8bf h[2]; } u;
  u.h[0] = *(const v8bf*)(p);
  u.h[1] = *(const v8bf*)(p + 8);
  return u.f;
}

// B-fragment (32x16) from BT (row-major B-transpose, i.e. column-major B):
// lane<16 -> col n0+lane, K 0..15 contiguous; lane>=16 -> K 16..31
__device__ __forceinline__ v16bf load_bt_frag(const bf16_t* __restrict__ BT, int ldb,
                                              int col0, int k0, int lane) {
  int n    = col0 + (lane & 15);
  int koff = (lane < 16) ? 0 : 16;
  return *(const v16bf*)(BT + (size_t)n * ldb + k0 + koff);
}

// ---------------- conversion kernels ----------------
__global__ void cvt_kernel(const float* __restrict__ in, bf16_t* __restrict__ out, int n) {
  int i = blockIdx.x * blockDim.x + threadIdx.x;
  if (i < n) out[i] = f2bf(in[i]);
}

__global__ void cvt_t_kernel(const float* __restrict__ in, bf16_t* __restrict__ out,
                             int K, int N) {
  int i = blockIdx.x * blockDim.x + threadIdx.x;
  if (i < K * N) {
    int k = i / N, n = i % N;
    out[(size_t)n * K + k] = f2bf(in[i]);
  }
}

// ---------------- fused QKV GEMM ------------------------------------------
// Wave tile: 32(M) x 64(N) = 8 accumulators; 2 A-frags + 4 B-frags per k-step
// feed 8 WMMAs (1.5 B128 loads per WMMA). Block = 8 waves as 2(M) x 4(N).
// Epilogue scatters into Q,K [B,H,T,D] and V^T [B,H,D,T].
__global__ __launch_bounds__(256) void gemm_qkv_kernel(const bf16_t* __restrict__ A,
                                                       const bf16_t* __restrict__ BT,
                                                       bf16_t* __restrict__ Qo,
                                                       bf16_t* __restrict__ Ko,
                                                       bf16_t* __restrict__ VTo) {
  const int Kd = CC;
  int wave = threadIdx.x >> 5, lane = threadIdx.x & 31;
  int wm = wave & 1, wn = wave >> 1;
  int m0 = blockIdx.y * 64 + wm * 32;
  int n0 = blockIdx.x * 256 + wn * 64;
  v8f c[2][4] = {};
  for (int k0 = 0; k0 < Kd; k0 += 32) {
    v16bf a0 = load_a_frag(A, Kd, m0,      k0, lane);
    v16bf a1 = load_a_frag(A, Kd, m0 + 16, k0, lane);
    v16bf b0 = load_bt_frag(BT, Kd, n0,      k0, lane);
    v16bf b1 = load_bt_frag(BT, Kd, n0 + 16, k0, lane);
    v16bf b2 = load_bt_frag(BT, Kd, n0 + 32, k0, lane);
    v16bf b3 = load_bt_frag(BT, Kd, n0 + 48, k0, lane);
    c[0][0] = wmma_bf16(a0, b0, c[0][0]);
    c[0][1] = wmma_bf16(a0, b1, c[0][1]);
    c[0][2] = wmma_bf16(a0, b2, c[0][2]);
    c[0][3] = wmma_bf16(a0, b3, c[0][3]);
    c[1][0] = wmma_bf16(a1, b0, c[1][0]);
    c[1][1] = wmma_bf16(a1, b1, c[1][1]);
    c[1][2] = wmma_bf16(a1, b2, c[1][2]);
    c[1][3] = wmma_bf16(a1, b3, c[1][3]);
  }
  int hi = lane >> 4, ln = lane & 15;
#pragma unroll
  for (int jn = 0; jn < 4; ++jn) {
    int n = n0 + jn * 16 + ln;
    int which = n / CC;
    int ccol  = n % CC;
    int h = ccol / DD, d = ccol % DD;
#pragma unroll
    for (int im = 0; im < 2; ++im) {
#pragma unroll
      for (int i = 0; i < 8; ++i) {
        int m = m0 + im * 16 + hi * 8 + i;
        int bb = m / TT, t = m % TT;
        bf16_t val = f2bf(c[im][jn][i]);
        if (which == 0)      Qo[(((size_t)bb * HH + h) * TT + t) * DD + d] = val;
        else if (which == 1) Ko[(((size_t)bb * HH + h) * TT + t) * DD + d] = val;
        else                 VTo[(((size_t)bb * HH + h) * DD + d) * TT + t] = val;
      }
    }
  }
}

// ---------------- flash-style causal attention ----------------
__global__ __launch_bounds__(256) void attn_kernel(const bf16_t* __restrict__ Q,
                                                   const bf16_t* __restrict__ Km,
                                                   const bf16_t* __restrict__ VT,
                                                   bf16_t* __restrict__ O) {
  __shared__ __align__(32) bf16_t plds[8][16][32];   // per-wave P tile (16 q x 32 k)
  int wave = threadIdx.x >> 5;
  int lane = threadIdx.x & 31;
  int gid  = blockIdx.x * 8 + wave;
  int bh = gid >> 7;             // 128 q-tiles per (b,h)
  int qb = (gid & 127) << 4;
  int b = bh >> 4, h = bh & 15;
  const bf16_t* Qb = Q  + (size_t)bh * TT * DD;
  const bf16_t* Kb = Km + (size_t)bh * TT * DD;
  const bf16_t* Vb = VT + (size_t)bh * DD * TT;
  int hi = lane >> 4, ln = lane & 15;

  v16bf qf0 = load_a_frag(Qb, DD, qb, 0, lane);
  v16bf qf1 = load_a_frag(Qb, DD, qb, 32, lane);

  v8f acc0 = {}, acc1 = {}, acc2 = {}, acc3 = {};
  float mrow[8], lrow[8];
#pragma unroll
  for (int i = 0; i < 8; ++i) { mrow[i] = -1e30f; lrow[i] = 0.f; }

  const float sc = 0.125f;                // 1/sqrt(64)
  int kend = qb + 16;
  for (int kb = 0; kb < kend; kb += 32) {
    v8f s0 = {}, s1 = {};
    {
      const bf16_t* kp = Kb + (size_t)(kb + ln) * DD + hi * 16;
      s0 = wmma_bf16(qf0, *(const v16bf*)(kp), s0);
      s0 = wmma_bf16(qf1, *(const v16bf*)(kp + 32), s0);
      const bf16_t* kp2 = Kb + (size_t)(kb + 16 + ln) * DD + hi * 16;
      s1 = wmma_bf16(qf0, *(const v16bf*)(kp2), s1);
      s1 = wmma_bf16(qf1, *(const v16bf*)(kp2 + 32), s1);
    }
    float p0[8], p1[8], rmax[8];
#pragma unroll
    for (int i = 0; i < 8; ++i) {
      int qg = qb + hi * 8 + i;
      float v0 = (kb + ln      <= qg) ? s0[i] * sc : -1e30f;
      float v1 = (kb + 16 + ln <= qg) ? s1[i] * sc : -1e30f;
      p0[i] = v0; p1[i] = v1;
      float mx = fmaxf(v0, v1);
      mx = fmaxf(mx, __shfl_xor(mx, 1));
      mx = fmaxf(mx, __shfl_xor(mx, 2));
      mx = fmaxf(mx, __shfl_xor(mx, 4));
      mx = fmaxf(mx, __shfl_xor(mx, 8));
      rmax[i] = mx;
    }
#pragma unroll
    for (int i = 0; i < 8; ++i) {
      float mn    = fmaxf(mrow[i], rmax[i]);
      float alpha = __expf(mrow[i] - mn);
      mrow[i] = mn;
      float e0 = __expf(p0[i] - mn);
      float e1 = __expf(p1[i] - mn);
      float s  = e0 + e1;
      s += __shfl_xor(s, 1); s += __shfl_xor(s, 2);
      s += __shfl_xor(s, 4); s += __shfl_xor(s, 8);
      lrow[i] = lrow[i] * alpha + s;
      acc0[i] *= alpha; acc1[i] *= alpha; acc2[i] *= alpha; acc3[i] *= alpha;
      plds[wave][hi * 8 + i][ln]      = f2bf(e0);
      plds[wave][hi * 8 + i][16 + ln] = f2bf(e1);
    }
    __builtin_amdgcn_wave_barrier();
    union { v16bf f; v8bf hh[2]; } pu;
    const bf16_t* pr = &plds[wave][ln][hi * 16];
    pu.hh[0] = *(const v8bf*)(pr);
    pu.hh[1] = *(const v8bf*)(pr + 8);
    __builtin_amdgcn_wave_barrier();
    v16bf pf = pu.f;
#pragma unroll
    for (int t = 0; t < 4; ++t) {
      const bf16_t* vp = Vb + (size_t)(t * 16 + ln) * TT + kb + hi * 16;
      v16bf vf = *(const v16bf*)(vp);
      v8f& a = (t == 0) ? acc0 : (t == 1) ? acc1 : (t == 2) ? acc2 : acc3;
      a = wmma_bf16(pf, vf, a);
    }
  }
#pragma unroll
  for (int i = 0; i < 8; ++i) {
    float inv = 1.0f / lrow[i];
    int qg = qb + hi * 8 + i;
    size_t base = ((size_t)b * TT + qg) * CC + (size_t)h * DD;
    O[base + 0  + ln] = f2bf(acc0[i] * inv);
    O[base + 16 + ln] = f2bf(acc1[i] * inv);
    O[base + 32 + ln] = f2bf(acc2[i] * inv);
    O[base + 48 + ln] = f2bf(acc3[i] * inv);
  }
}

// ---------------- output projection GEMM (f32 out) ----------------
// Same 32x64 register blocking as the QKV GEMM.
__global__ __launch_bounds__(256) void gemm_proj_kernel(const bf16_t* __restrict__ A,
                                                        const bf16_t* __restrict__ BT,
                                                        float* __restrict__ Out) {
  const int Kd = CC, Nn = CC;
  int wave = threadIdx.x >> 5, lane = threadIdx.x & 31;
  int wm = wave & 1, wn = wave >> 1;
  int m0 = blockIdx.y * 64 + wm * 32;
  int n0 = blockIdx.x * 256 + wn * 64;
  v8f c[2][4] = {};
  for (int k0 = 0; k0 < Kd; k0 += 32) {
    v16bf a0 = load_a_frag(A, Kd, m0,      k0, lane);
    v16bf a1 = load_a_frag(A, Kd, m0 + 16, k0, lane);
    v16bf b0 = load_bt_frag(BT, Kd, n0,      k0, lane);
    v16bf b1 = load_bt_frag(BT, Kd, n0 + 16, k0, lane);
    v16bf b2 = load_bt_frag(BT, Kd, n0 + 32, k0, lane);
    v16bf b3 = load_bt_frag(BT, Kd, n0 + 48, k0, lane);
    c[0][0] = wmma_bf16(a0, b0, c[0][0]);
    c[0][1] = wmma_bf16(a0, b1, c[0][1]);
    c[0][2] = wmma_bf16(a0, b2, c[0][2]);
    c[0][3] = wmma_bf16(a0, b3, c[0][3]);
    c[1][0] = wmma_bf16(a1, b0, c[1][0]);
    c[1][1] = wmma_bf16(a1, b1, c[1][1]);
    c[1][2] = wmma_bf16(a1, b2, c[1][2]);
    c[1][3] = wmma_bf16(a1, b3, c[1][3]);
  }
  int hi = lane >> 4, ln = lane & 15;
#pragma unroll
  for (int jn = 0; jn < 4; ++jn) {
    int n = n0 + jn * 16 + ln;
#pragma unroll
    for (int im = 0; im < 2; ++im) {
#pragma unroll
      for (int i = 0; i < 8; ++i) {
        int m = m0 + im * 16 + hi * 8 + i;
        Out[(size_t)m * Nn + n] = c[im][jn][i];
      }
    }
  }
}

extern "C" void kernel_launch(void* const* d_in, const int* in_sizes, int n_in,
                              void* d_out, int out_size, void* d_ws, size_t ws_size,
                              hipStream_t stream) {
  (void)in_sizes; (void)n_in; (void)out_size; (void)ws_size;
  const float* x    = (const float*)d_in[0];
  const float* wqkv = (const float*)d_in[1];
  const float* wout = (const float*)d_in[2];
  float* out = (float*)d_out;

  char* ws = (char*)d_ws;
  const size_t MB = 1024 * 1024;
  bf16_t* xb    = (bf16_t*)(ws + 0);        // [4096,1024]      8 MiB
  bf16_t* wqkvT = (bf16_t*)(ws + 8 * MB);   // [3072,1024]      6 MiB
  bf16_t* woutT = (bf16_t*)(ws + 14 * MB);  // [1024,1024]      2 MiB
  bf16_t* Qb    = (bf16_t*)(ws + 16 * MB);  // [B*H,T,D]        8 MiB
  bf16_t* Kb    = (bf16_t*)(ws + 24 * MB);  // [B*H,T,D]        8 MiB
  bf16_t* VTb   = (bf16_t*)(ws + 32 * MB);  // [B*H,D,T]        8 MiB
  bf16_t* AOb   = (bf16_t*)(ws + 40 * MB);  // [4096,1024]      8 MiB

  cvt_kernel<<<(BB * TT * CC) / 256, 256, 0, stream>>>(x, xb, BB * TT * CC);
  cvt_t_kernel<<<(CC * 3 * CC) / 256, 256, 0, stream>>>(wqkv, wqkvT, CC, 3 * CC);
  cvt_t_kernel<<<(CC * CC) / 256, 256, 0, stream>>>(wout, woutT, CC, CC);

  dim3 gq(3 * CC / 256, BB * TT / 64);   // (12, 64)
  gemm_qkv_kernel<<<gq, 256, 0, stream>>>(xb, wqkvT, Qb, Kb, VTb);

  attn_kernel<<<(BB * HH * (TT / 16)) / 8, 256, 0, stream>>>(Qb, Kb, VTb, AOb);

  dim3 gp(CC / 256, BB * TT / 64);       // (4, 64)
  gemm_proj_kernel<<<gp, 256, 0, stream>>>(AOb, woutT, out);
}
